// CausalSelfAttention_72284299592389
// MI455X (gfx1250) — matrix operand order repaired and checked by reading, not compiled, not run on hardware
//
#include <hip/hip_runtime.h>

// ---------------------------------------------------------------------------
// CDNA5 (gfx1250) causal self-attention forward:
//   qkv = x @ Wqkv^T + b ; flash-attention per head ; out = attn @ Wproj^T + b
// All GEMM work on V_WMMA_F32_16X16X32_F16 (f16 in, f32 accumulate).
// K-tile staging uses GLOBAL_LOAD_ASYNC_TO_LDS_B128 (ASYNCcnt path).
// ---------------------------------------------------------------------------

typedef __attribute__((ext_vector_type(16))) _Float16 v16h;
typedef __attribute__((ext_vector_type(8)))  float    v8f;

union AF { v16h v; uint4 u[2]; };

static __device__ __forceinline__ v8f v8f_zero() {
    v8f z = {0.f, 0.f, 0.f, 0.f, 0.f, 0.f, 0.f, 0.f};
    return z;
}

#define WMMA_F16(a, b, c) \
    __builtin_amdgcn_wmma_f32_16x16x32_f16(false, (a), false, (b), (short)0, (c), false, false)

constexpr int T_ = 2048;   // tokens
constexpr int C_ = 1024;   // embed dim
constexpr int H_ = 16;     // heads
constexpr int D_ = 64;     // head dim

// ---------------------------------------------------------------------------
// fp32 -> f16 convert
// ---------------------------------------------------------------------------
__global__ void cvt_f32_to_f16(const float* __restrict__ in, _Float16* __restrict__ out, int n) {
    int i = blockIdx.x * blockDim.x + threadIdx.x;
    if (i < n) out[i] = (_Float16)in[i];
}

// ---------------------------------------------------------------------------
// GEMM: Y[M,N] = X[M,K] * W[N,K]^T + bias[N]
// X, W row-major f16. 256 threads = 8 waves (2 M x 4 N); each wave does a
// 16x64 strip (4 WMMA accumulators). M%32==0, N%256==0, K%32==0 assumed.
//
// WMMA f16 fragment layouts (CDNA5 ISA 7.12.2):
//   A 16x32: lane<16 row=lane, halfs h -> K = h  (this code's contiguous order)
//            lane>=16 row=lane-16,    K = 16 + h
//   B 32x16: lane: N = lane%16, K = (lane<16 ? 0 : 16) + h   (16 contiguous, this code's order as for A)
//   C/D f32 16x16: vgpr i: lane<16 -> (M=i, N=lane); lane>=16 -> (M=i+8, N=lane-16)
// ---------------------------------------------------------------------------
template <bool OUT16>
__global__ __launch_bounds__(256) void gemm_bias_wmma(const _Float16* __restrict__ X,
                                                      const _Float16* __restrict__ W,
                                                      const float* __restrict__ bias,
                                                      void* __restrict__ Yv,
                                                      int M, int N, int K) {
    const int wave   = threadIdx.x >> 5;
    const int lane   = threadIdx.x & 31;
    const int lane16 = lane & 15;
    const int hi     = lane >> 4;           // 0 or 1
    const int waveM  = wave >> 2;           // 0..1
    const int waveN  = wave & 3;            // 0..3
    const int rowBase = blockIdx.y * 32 + waveM * 16;
    const int colBase = blockIdx.x * 256 + waveN * 64;

    const _Float16* xrow  = X + (size_t)(rowBase + lane16) * K;
    const _Float16* wrow0 = W + (size_t)(colBase +  0 + lane16) * K;
    const _Float16* wrow1 = W + (size_t)(colBase + 16 + lane16) * K;
    const _Float16* wrow2 = W + (size_t)(colBase + 32 + lane16) * K;
    const _Float16* wrow3 = W + (size_t)(colBase + 48 + lane16) * K;
    const int offA = hi ? 16 : 0;   // A-frag K offset for this lane half
    const int offB = hi ? 16 : 0;   // B-frag K offset for this lane half

    v8f acc0 = v8f_zero(), acc1 = v8f_zero(), acc2 = v8f_zero(), acc3 = v8f_zero();

#pragma unroll 2
    for (int kb = 0; kb < K; kb += 32) {
        if (kb + 64 < K) {
            __builtin_prefetch((const void*)(xrow + kb + 64), 0, 3);
            __builtin_prefetch((const void*)(wrow0 + kb + 64), 0, 3);
            __builtin_prefetch((const void*)(wrow1 + kb + 64), 0, 3);
            __builtin_prefetch((const void*)(wrow2 + kb + 64), 0, 3);
            __builtin_prefetch((const void*)(wrow3 + kb + 64), 0, 3);
        }
        AF a;
        a.u[0] = *(const uint4*)(xrow + kb + offA);
        a.u[1] = *(const uint4*)(xrow + kb + offA + 8);
        const int cb = kb + offB;
        AF b0, b1, b2, b3;
        b0.u[0] = *(const uint4*)(wrow0 + cb); b0.u[1] = *(const uint4*)(wrow0 + cb + 8);
        b1.u[0] = *(const uint4*)(wrow1 + cb); b1.u[1] = *(const uint4*)(wrow1 + cb + 8);
        b2.u[0] = *(const uint4*)(wrow2 + cb); b2.u[1] = *(const uint4*)(wrow2 + cb + 8);
        b3.u[0] = *(const uint4*)(wrow3 + cb); b3.u[1] = *(const uint4*)(wrow3 + cb + 8);
        acc0 = WMMA_F16(a.v, b0.v, acc0);
        acc1 = WMMA_F16(a.v, b1.v, acc1);
        acc2 = WMMA_F16(a.v, b2.v, acc2);
        acc3 = WMMA_F16(a.v, b3.v, acc3);
    }

    // Epilogue: bias add + store (C/D layout -> row major)
    const int rowOff = rowBase + (hi ? 8 : 0);
    v8f accs[4] = {acc0, acc1, acc2, acc3};
#pragma unroll
    for (int t = 0; t < 4; ++t) {
        const int n = colBase + t * 16 + lane16;
        const float bv = bias[n];
#pragma unroll
        for (int i = 0; i < 8; ++i) {
            const int r = rowOff + i;
            const float y = accs[t][i] + bv;
            if constexpr (OUT16) ((_Float16*)Yv)[(size_t)r * N + n] = (_Float16)y;
            else                 ((float*)Yv)[(size_t)r * N + n] = y;
        }
    }
}

// ---------------------------------------------------------------------------
// Flash attention (causal). qkv: [T, 3C] f16 (Q | K | V per row); out: [T, C] f16.
// Block = 128 threads = 4 waves; block covers 64 query rows of one head;
// each wave owns 16 query rows with a full 16xD accumulator (online softmax,
// wave-private -> no cross-wave reductions). Key blocks of 64 so that one
// softmax pass (shuffle-reduces + O rescale) amortizes over 16 WMMAs.
// K tile staged with async global->LDS; V tile transposed via packed b32 stores.
// ---------------------------------------------------------------------------
__global__ __launch_bounds__(128) void attn_wmma(const _Float16* __restrict__ qkv,
                                                 _Float16* __restrict__ out) {
    __shared__ __align__(16) _Float16 Ks[64][80];       // [key][d]        (160B rows)
    __shared__ __align__(16) _Float16 Vt[64][80];       // [d][key] transp (160B rows)
    __shared__ __align__(16) _Float16 Pb[4][16][64];    // per-wave P relayout buffer

    const int wave   = threadIdx.x >> 5;
    const int lane   = threadIdx.x & 31;
    const int lane16 = lane & 15;
    const int hi     = lane >> 4;
    const int head   = blockIdx.y;
    const int q0     = blockIdx.x * 64;
    const int qrow   = q0 + wave * 16;
    const int stride = 3 * C_;
    const int hcol   = head * D_;
    const int offA   = hi ? 16 : 0;
    const int offB   = hi ? 16 : 0;
    const int koff   = hi ? 16 : 0;

    // Q fragments for this wave's 16 rows: A-layout, K-dim = d (two k=32 frags)
    const _Float16* qp = qkv + (size_t)(qrow + lane16) * stride + hcol;
    AF qa0, qa1;
    qa0.u[0] = *(const uint4*)(qp + offA);
    qa0.u[1] = *(const uint4*)(qp + offA + 8);
    qa1.u[0] = *(const uint4*)(qp + 32 + offA);
    qa1.u[1] = *(const uint4*)(qp + 32 + offA + 8);

    v8f O[4];
#pragma unroll
    for (int t = 0; t < 4; ++t) O[t] = v8f_zero();
    float mrow[8], lrow[8];
#pragma unroll
    for (int i = 0; i < 8; ++i) { mrow[i] = -1e30f; lrow[i] = 0.f; }

    const float scale = 0.125f;               // 1/sqrt(64)
    const int nkb = blockIdx.x + 1;           // 64-key blocks (uniform across block)

    for (int kb = 0; kb < nkb; ++kb) {
        const int kt = kb * 64;
        __syncthreads();

        // ---- K tile: async global -> LDS (no VGPR bounce, ASYNCcnt) ----
        {
            const int tt = threadIdx.x;
#pragma unroll
            for (int j = 0; j < 4; ++j) {
                const int e  = tt + 128 * j;      // 16B chunk id, 0..511 over 64x64 tile
                const int kk = e >> 3;            // key row
                const int c0 = (e & 7) * 8;       // d column (halfs)
                const _Float16* gp = qkv + (size_t)(kt + kk) * stride + C_ + hcol + c0;
                const unsigned lds = (unsigned)(uintptr_t)&Ks[kk][c0];
                asm volatile("global_load_async_to_lds_b128 %0, %1, off"
                             :
                             : "v"(lds), "v"((unsigned long long)(uintptr_t)gp)
                             : "memory");
            }
        }
        // ---- V tile: load two adjacent keys, transpose, packed b32 LDS stores ----
        {
            const int tt = threadIdx.x;
            const int a  = tt >> 2;            // key pair -> keys {2a, 2a+1}
            const int c0 = (tt & 3) * 16;      // 0,16,32,48
            const _Float16* vp0 = qkv + (size_t)(kt + 2 * a)     * stride + 2 * C_ + hcol + c0;
            const _Float16* vp1 = qkv + (size_t)(kt + 2 * a + 1) * stride + 2 * C_ + hcol + c0;
            union { uint4 u[2]; unsigned short h[16]; } r0, r1;
            r0.u[0] = *(const uint4*)(vp0);  r0.u[1] = *(const uint4*)(vp0 + 8);
            r1.u[0] = *(const uint4*)(vp1);  r1.u[1] = *(const uint4*)(vp1 + 8);
#pragma unroll
            for (int j = 0; j < 16; ++j) {
                const unsigned pk = (unsigned)r0.h[j] | ((unsigned)r1.h[j] << 16);
                *(unsigned*)&Vt[c0 + j][2 * a] = pk;
            }
        }
        asm volatile("s_wait_asynccnt 0x0" ::: "memory");
        __syncthreads();

        // ---- S = Q K^T : four 16x16 tiles (key groups of 16) ----
        v8f S[4];
#pragma unroll
        for (int g = 0; g < 4; ++g) {
            S[g] = v8f_zero();
            const int key = g * 16 + lane16;
            AF b;
            b.u[0] = *(const uint4*)&Ks[key][offB];
            b.u[1] = *(const uint4*)&Ks[key][offB + 8];
            S[g] = WMMA_F16(qa0.v, b.v, S[g]);
            b.u[0] = *(const uint4*)&Ks[key][32 + offB];
            b.u[1] = *(const uint4*)&Ks[key][32 + offB + 8];
            S[g] = WMMA_F16(qa1.v, b.v, S[g]);
        }

        // ---- scale, causal mask, online softmax (rows span 16-lane groups) ----
#pragma unroll
        for (int i = 0; i < 8; ++i) {
            const int M = i + (hi ? 8 : 0);
            const int q = qrow + M;
            float s[4], p[4];
            bool ok[4];
#pragma unroll
            for (int g = 0; g < 4; ++g) {
                ok[g] = (kt + g * 16 + lane16) <= q;
                s[g]  = ok[g] ? S[g][i] * scale : -1e30f;
            }
            float mx = fmaxf(fmaxf(s[0], s[1]), fmaxf(s[2], s[3]));
            mx = fmaxf(mx, __shfl_xor(mx, 1, 32));
            mx = fmaxf(mx, __shfl_xor(mx, 2, 32));
            mx = fmaxf(mx, __shfl_xor(mx, 4, 32));
            mx = fmaxf(mx, __shfl_xor(mx, 8, 32));
            const float mnew  = fmaxf(mrow[i], mx);
            const float alpha = __expf(mrow[i] - mnew);
            float ps = 0.f;
#pragma unroll
            for (int g = 0; g < 4; ++g) {
                p[g] = ok[g] ? __expf(s[g] - mnew) : 0.f;
                ps += p[g];
            }
            ps += __shfl_xor(ps, 1, 32);
            ps += __shfl_xor(ps, 2, 32);
            ps += __shfl_xor(ps, 4, 32);
            ps += __shfl_xor(ps, 8, 32);
            lrow[i] = lrow[i] * alpha + ps;
            mrow[i] = mnew;
#pragma unroll
            for (int t = 0; t < 4; ++t) O[t][i] *= alpha;
#pragma unroll
            for (int g = 0; g < 4; ++g) Pb[wave][M][g * 16 + lane16] = (_Float16)p[g];
        }
        // wave-private LDS round-trip: per-wave DS ordering + explicit wait
        asm volatile("s_wait_dscnt 0x0" ::: "memory");

        // ---- O += P V : reload P as two A-fragments, V^T rows give B-frags ----
        AF pa0, pa1;
        pa0.u[0] = *(const uint4*)&Pb[wave][lane16][offA];
        pa0.u[1] = *(const uint4*)&Pb[wave][lane16][offA + 8];
        pa1.u[0] = *(const uint4*)&Pb[wave][lane16][32 + offA];
        pa1.u[1] = *(const uint4*)&Pb[wave][lane16][32 + offA + 8];
#pragma unroll
        for (int t = 0; t < 4; ++t) {
            AF vb;
            const int dcol = t * 16 + lane16;
            vb.u[0] = *(const uint4*)&Vt[dcol][koff];
            vb.u[1] = *(const uint4*)&Vt[dcol][koff + 8];
            O[t] = WMMA_F16(pa0.v, vb.v, O[t]);
            vb.u[0] = *(const uint4*)&Vt[dcol][32 + koff];
            vb.u[1] = *(const uint4*)&Vt[dcol][32 + koff + 8];
            O[t] = WMMA_F16(pa1.v, vb.v, O[t]);
        }
    }

    // ---- normalize and write attn output [T, C] f16 ----
#pragma unroll
    for (int i = 0; i < 8; ++i) {
        const int r = qrow + i + (hi ? 8 : 0);
        const float inv = 1.f / lrow[i];
#pragma unroll
        for (int t = 0; t < 4; ++t) {
            out[(size_t)r * C_ + hcol + t * 16 + lane16] = (_Float16)(O[t][i] * inv);
        }
    }
}

// ---------------------------------------------------------------------------
// Launch: converts -> QKV GEMM -> flash attention -> proj GEMM
// Workspace (f16): xh 2M | wqkvh 3M | projwh 1M | qkvh 6M | attnh 2M = 28 MB
// ---------------------------------------------------------------------------
extern "C" void kernel_launch(void* const* d_in, const int* in_sizes, int n_in,
                              void* d_out, int out_size, void* d_ws, size_t ws_size,
                              hipStream_t stream) {
    const float* x      = (const float*)d_in[0];
    const float* wqkv_w = (const float*)d_in[1];
    const float* wqkv_b = (const float*)d_in[2];
    const float* proj_w = (const float*)d_in[3];
    const float* proj_b = (const float*)d_in[4];
    float* out = (float*)d_out;

    _Float16* xh     = (_Float16*)d_ws;
    _Float16* wqkvh  = xh + (size_t)T_ * C_;
    _Float16* projwh = wqkvh + (size_t)3 * C_ * C_;
    _Float16* qkvh   = projwh + (size_t)C_ * C_;
    _Float16* attnh  = qkvh + (size_t)T_ * 3 * C_;

    const int n1 = T_ * C_;
    const int n2 = 3 * C_ * C_;
    const int n3 = C_ * C_;
    cvt_f32_to_f16<<<(n1 + 255) / 256, 256, 0, stream>>>(x, xh, n1);
    cvt_f32_to_f16<<<(n2 + 255) / 256, 256, 0, stream>>>(wqkv_w, wqkvh, n2);
    cvt_f32_to_f16<<<(n3 + 255) / 256, 256, 0, stream>>>(proj_w, projwh, n3);

    gemm_bias_wmma<true><<<dim3((3 * C_) / 256, T_ / 32), 256, 0, stream>>>(
        xh, wqkvh, wqkv_b, (void*)qkvh, T_, 3 * C_, C_);

    attn_wmma<<<dim3(T_ / 64, H_), 128, 0, stream>>>(qkvh, attnh);

    gemm_bias_wmma<false><<<dim3(C_ / 256, T_ / 32), 256, 0, stream>>>(
        attnh, projwh, proj_b, (void*)out, T_, C_, C_);
}
